// RelativeMultiHeadAttention_24670292149054
// MI455X (gfx1250) — matrix operand order repaired and checked by reading, not compiled, not run on hardware
//
#include <hip/hip_runtime.h>

// ---------------------------------------------------------------------------
// Relative MHA for MI455X (gfx1250, wave32, WMMA bf16 16x16x32, f32 accum).
// B=2, T=2048, D=1024, H=16, dk=64, MAX_REL=32.  q = k = v = x @ W_q^T.
// rel_logits[i,j] = q[i] . rel_k_emb[clip(i-j,-32,32)+32]  -> 16x65 table/tile.
// ---------------------------------------------------------------------------

#define T_SEQ   2048
#define DMODEL  1024
#define NHEADS  16
#define DK      64
#define MAXREL  32
#define NREL    65
#define BATCH   2
#define MTOT    (BATCH * T_SEQ)   // 4096 rows

typedef __attribute__((ext_vector_type(16))) __bf16 v16bf;
typedef __attribute__((ext_vector_type(8)))  float  v8f;

union FragU { uint4 u[2]; v16bf v; };

__device__ __forceinline__ unsigned short f2bf(float x) {
  unsigned u = __float_as_uint(x);
  u += 0x7fffu + ((u >> 16) & 1u);          // round-to-nearest-even
  return (unsigned short)(u >> 16);
}
__device__ __forceinline__ float bf2f(unsigned short h) {
  return __uint_as_float(((unsigned)h) << 16);
}

// A fragment: 16x32 (MxK) bf16, row-major source, ld in elements.
// ISA 7.12.2: lanes 0-15 (m) hold K 0..7 (v0-3) and 16..23 (v4-7);
//             lanes 16-31 hold K 8..15 and 24..31.
__device__ __forceinline__ v16bf load_frag_a(const __bf16* src, int ld) {
  int lane = threadIdx.x & 31;
  int m = lane & 15, kh = lane >> 4;
  FragU f;
  f.u[0] = *(const uint4*)(src + (size_t)m * ld + kh * 8);
  f.u[1] = *(const uint4*)(src + (size_t)m * ld + 16 + kh * 8);
  return f.v;
}

// B fragment: 32x16 (KxN).  src holds B^T row-major (row n == column n of B).
// lanes 0-15 (n) hold K 0..7 and 16..23, lanes 16-31 hold K 8..15 and 24..31.
__device__ __forceinline__ v16bf load_frag_b(const __bf16* src, int ld) {
  int lane = threadIdx.x & 31;
  int n = lane & 15, kh = lane >> 4;
  FragU f;
  f.u[0] = *(const uint4*)(src + (size_t)n * ld + kh * 8);
  f.u[1] = *(const uint4*)(src + (size_t)n * ld + 16 + kh * 8);
  return f.v;
}

__device__ __forceinline__ v8f wmma_bf16(v16bf a, v16bf b, v8f c) {
  return __builtin_amdgcn_wmma_f32_16x16x32_bf16(
      false, a, false, b, (short)0, c, false, false);
}

// ---------------------------------------------------------------------------
// fp32 -> bf16 elementwise convert
// ---------------------------------------------------------------------------
__global__ void cvt_bf16_kernel(const float* __restrict__ in,
                                unsigned short* __restrict__ out, int n) {
  int i = blockIdx.x * 256 + threadIdx.x;
  if (i < n) out[i] = f2bf(in[i]);
}

// ---------------------------------------------------------------------------
// Q projection: Q = Xb @ Wqb^T  (M=4096, N=1024, K=1024)
// Wave tile 32x64, WG 8 waves (4x2) -> 128x128 per block.
// Writes Qh[bh][t][dk] and Qt[bh][dk][t] in bf16.
// ---------------------------------------------------------------------------
__global__ __launch_bounds__(256) void qproj_kernel(
    const unsigned short* __restrict__ Xb,
    const unsigned short* __restrict__ Wqb,
    unsigned short* __restrict__ Qh,
    unsigned short* __restrict__ Qt) {
  const int tid  = threadIdx.x;
  const int w    = tid >> 5;
  const int lane = tid & 31;
  const int m0 = blockIdx.x * 128 + (w & 3) * 32;
  const int n0 = blockIdx.y * 128 + (w >> 2) * 64;

  v8f zero = {0.f, 0.f, 0.f, 0.f, 0.f, 0.f, 0.f, 0.f};
  v8f acc[2][4];
#pragma unroll
  for (int mt = 0; mt < 2; ++mt)
#pragma unroll
    for (int nt = 0; nt < 4; ++nt) acc[mt][nt] = zero;

  for (int kk = 0; kk < DMODEL; kk += 32) {
    v16bf a[2], b[4];
#pragma unroll
    for (int mt = 0; mt < 2; ++mt)
      a[mt] = load_frag_a((const __bf16*)Xb + (size_t)(m0 + mt * 16) * DMODEL + kk, DMODEL);
#pragma unroll
    for (int nt = 0; nt < 4; ++nt)
      b[nt] = load_frag_b((const __bf16*)Wqb + (size_t)(n0 + nt * 16) * DMODEL + kk, DMODEL);
#pragma unroll
    for (int mt = 0; mt < 2; ++mt)
#pragma unroll
      for (int nt = 0; nt < 4; ++nt)
        acc[mt][nt] = wmma_bf16(a[mt], b[nt], acc[mt][nt]);
  }

  const int hl = lane >> 4, nn = lane & 15;
#pragma unroll
  for (int mt = 0; mt < 2; ++mt)
#pragma unroll
    for (int nt = 0; nt < 4; ++nt)
#pragma unroll
      for (int r = 0; r < 8; ++r) {
        int gm = m0 + mt * 16 + r + 8 * hl;   // row in [0, 4096): b*T + t
        int gn = n0 + nt * 16 + nn;           // col in [0, 1024): h*64 + d
        unsigned short hv = f2bf(acc[mt][nt][r]);
        int bb = gm >> 11, t = gm & 2047;
        int h  = gn >> 6,  d = gn & 63;
        int bh = bb * NHEADS + h;
        Qh[(size_t)(bh * T_SEQ + t) * DK + d] = hv;
        Qt[(size_t)(bh * DK + d) * T_SEQ + t] = hv;
      }
}

// ---------------------------------------------------------------------------
// Flash attention with relative-position bias.
// One wave owns a 16-row i-tile of one (b,h).  Online softmax over j in
// blocks of 64.  P is converted C-layout -> A-layout through wave-private LDS.
// ---------------------------------------------------------------------------
__global__ __launch_bounds__(256) void attn_kernel(
    const unsigned short* __restrict__ Qh,
    const unsigned short* __restrict__ Qt,
    const float* __restrict__ relk,
    unsigned short* __restrict__ Ab) {
  __shared__ float relbuf[8][16][NREL + 3];                 // 16x65 per wave
  __shared__ __align__(16) unsigned short pbuf[8][16][64];  // P tile per wave

  const int tid  = threadIdx.x;
  const int w    = tid >> 5;
  const int lane = tid & 31;
  const int g  = blockIdx.x * 8 + w;   // global wave id, [0, 4096)
  const int bh = g >> 7;               // [0, 32)
  const int i0 = (g & 127) << 4;       // row tile origin

  // --- rel-position table: r[row][c] = q[i0+row] . rel_k_emb[c] ------------
  for (int idx = lane; idx < 16 * NREL; idx += 32) {
    int row = idx / NREL, c = idx - row * NREL;
    const unsigned short* qr = Qh + (size_t)(bh * T_SEQ + i0 + row) * DK;
    const float* rp = relk + c * DK;
    float acc = 0.f;
#pragma unroll 8
    for (int d = 0; d < DK; ++d) acc += bf2f(qr[d]) * rp[d];
    relbuf[w][row][c] = acc;
  }
  asm volatile("s_wait_dscnt 0x0" ::: "memory");

  // Q_i fragments stay resident (16x64 -> two K=32 A fragments).
  const __bf16* qbase = (const __bf16*)Qh + (size_t)(bh * T_SEQ + i0) * DK;
  v16bf aq0 = load_frag_a(qbase, DK);
  v16bf aq1 = load_frag_a(qbase + 32, DK);

  v8f zero = {0.f, 0.f, 0.f, 0.f, 0.f, 0.f, 0.f, 0.f};
  v8f o[4] = {zero, zero, zero, zero};
  float m8[8], l8[8];
#pragma unroll
  for (int r = 0; r < 8; ++r) { m8[r] = -3.0e38f; l8[r] = 0.f; }

  const int hl = lane >> 4, nn = lane & 15;
  const float scale = 0.125f;                 // 1/sqrt(64)
  const float LOG2E = 1.4426950408889634f;

  for (int j0 = 0; j0 < T_SEQ; j0 += 64) {
    // ---- S = Q_i @ K_j^T ---------------------------------------------------
    v8f s[4];
#pragma unroll
    for (int nt = 0; nt < 4; ++nt) {
      const __bf16* kb = (const __bf16*)Qh + (size_t)(bh * T_SEQ + j0 + nt * 16) * DK;
      v8f t0 = wmma_bf16(aq0, load_frag_b(kb, DK), zero);
      s[nt]  = wmma_bf16(aq1, load_frag_b(kb + 32, DK), t0);
    }
    // ---- rel bias + scale + row max ---------------------------------------
    float newm[8];
#pragma unroll
    for (int r = 0; r < 8; ++r) newm[r] = m8[r];
#pragma unroll
    for (int nt = 0; nt < 4; ++nt)
#pragma unroll
      for (int r = 0; r < 8; ++r) {
        int mrow = r + 8 * hl;
        int j = j0 + nt * 16 + nn;
        int dd = (i0 + mrow) - j;
        dd = dd < -MAXREL ? -MAXREL : (dd > MAXREL ? MAXREL : dd);
        float val = (s[nt][r] + relbuf[w][mrow][dd + MAXREL]) * scale;
        s[nt][r] = val;
        newm[r] = fmaxf(newm[r], val);
      }
#pragma unroll
    for (int r = 0; r < 8; ++r)
#pragma unroll
      for (int off = 1; off < 16; off <<= 1)
        newm[r] = fmaxf(newm[r], __shfl_xor(newm[r], off, 32));

    // ---- online softmax update --------------------------------------------
    float alpha[8], rsum[8];
#pragma unroll
    for (int r = 0; r < 8; ++r) {
      alpha[r] = exp2f((m8[r] - newm[r]) * LOG2E);
      m8[r] = newm[r];
      rsum[r] = 0.f;
    }
#pragma unroll
    for (int nt = 0; nt < 4; ++nt)
#pragma unroll
      for (int r = 0; r < 8; ++r) {
        float p = exp2f((s[nt][r] - m8[r]) * LOG2E);
        s[nt][r] = p;
        rsum[r] += p;
      }
#pragma unroll
    for (int r = 0; r < 8; ++r) {
#pragma unroll
      for (int off = 1; off < 16; off <<= 1)
        rsum[r] += __shfl_xor(rsum[r], off, 32);
      l8[r] = l8[r] * alpha[r] + rsum[r];
    }
#pragma unroll
    for (int ot = 0; ot < 4; ++ot)
#pragma unroll
      for (int r = 0; r < 8; ++r) o[ot][r] *= alpha[r];

    // ---- P: C-layout -> A-layout via wave-private LDS ---------------------
#pragma unroll
    for (int nt = 0; nt < 4; ++nt)
#pragma unroll
      for (int r = 0; r < 8; ++r)
        pbuf[w][r + 8 * hl][nt * 16 + nn] = f2bf(s[nt][r]);
    asm volatile("s_wait_dscnt 0x0" ::: "memory");

    // ---- O += P @ V  (V rows come from Qt so B loads are row loads) -------
#pragma unroll
    for (int kc = 0; kc < 2; ++kc) {
      v16bf pa = load_frag_a((const __bf16*)&pbuf[w][0][kc * 32], 64);
#pragma unroll
      for (int ot = 0; ot < 4; ++ot) {
        const __bf16* vb = (const __bf16*)Qt +
            (size_t)(bh * DK + ot * 16) * T_SEQ + j0 + kc * 32;
        o[ot] = wmma_bf16(pa, load_frag_b(vb, T_SEQ), o[ot]);
      }
    }
  }

  // ---- epilogue: O /= l, scatter into [B*T, D] bf16 -------------------------
  int bb = bh >> 4, h = bh & 15;
#pragma unroll
  for (int r = 0; r < 8; ++r) {
    float inv = 1.0f / l8[r];
    int mrow = r + 8 * hl;
    size_t rowbase = (size_t)(bb * T_SEQ + i0 + mrow) * DMODEL + h * DK;
#pragma unroll
    for (int ot = 0; ot < 4; ++ot)
      Ab[rowbase + ot * 16 + nn] = f2bf(o[ot][r] * inv);
  }
}

// ---------------------------------------------------------------------------
// Output projection: out = Ab @ Wob^T -> fp32 d_out  (M=4096, N=1024, K=1024)
// ---------------------------------------------------------------------------
__global__ __launch_bounds__(256) void oproj_kernel(
    const unsigned short* __restrict__ Ab,
    const unsigned short* __restrict__ Wob,
    float* __restrict__ out) {
  const int tid  = threadIdx.x;
  const int w    = tid >> 5;
  const int lane = tid & 31;
  const int m0 = blockIdx.x * 128 + (w & 3) * 32;
  const int n0 = blockIdx.y * 128 + (w >> 2) * 64;

  v8f zero = {0.f, 0.f, 0.f, 0.f, 0.f, 0.f, 0.f, 0.f};
  v8f acc[2][4];
#pragma unroll
  for (int mt = 0; mt < 2; ++mt)
#pragma unroll
    for (int nt = 0; nt < 4; ++nt) acc[mt][nt] = zero;

  for (int kk = 0; kk < DMODEL; kk += 32) {
    v16bf a[2], b[4];
#pragma unroll
    for (int mt = 0; mt < 2; ++mt)
      a[mt] = load_frag_a((const __bf16*)Ab + (size_t)(m0 + mt * 16) * DMODEL + kk, DMODEL);
#pragma unroll
    for (int nt = 0; nt < 4; ++nt)
      b[nt] = load_frag_b((const __bf16*)Wob + (size_t)(n0 + nt * 16) * DMODEL + kk, DMODEL);
#pragma unroll
    for (int mt = 0; mt < 2; ++mt)
#pragma unroll
      for (int nt = 0; nt < 4; ++nt)
        acc[mt][nt] = wmma_bf16(a[mt], b[nt], acc[mt][nt]);
  }

  const int hl = lane >> 4, nn = lane & 15;
#pragma unroll
  for (int mt = 0; mt < 2; ++mt)
#pragma unroll
    for (int nt = 0; nt < 4; ++nt)
#pragma unroll
      for (int r = 0; r < 8; ++r) {
        int gm = m0 + mt * 16 + r + 8 * hl;
        int gn = n0 + nt * 16 + nn;
        out[(size_t)gm * DMODEL + gn] = acc[mt][nt][r];
      }
}

// ---------------------------------------------------------------------------
// Launch
// ---------------------------------------------------------------------------
extern "C" void kernel_launch(void* const* d_in, const int* in_sizes, int n_in,
                              void* d_out, int out_size, void* d_ws, size_t ws_size,
                              hipStream_t stream) {
  const float* x    = (const float*)d_in[0];
  const float* Wq   = (const float*)d_in[1];
  const float* Wo   = (const float*)d_in[2];
  const float* relk = (const float*)d_in[3];

  char* ws = (char*)d_ws;
  unsigned short* Xb  = (unsigned short*)(ws + (size_t)0);          // 8 MB
  unsigned short* Wqb = (unsigned short*)(ws + ((size_t)8  << 20)); // 2 MB
  unsigned short* Wob = (unsigned short*)(ws + ((size_t)10 << 20)); // 2 MB
  unsigned short* Qh  = (unsigned short*)(ws + ((size_t)12 << 20)); // 8 MB
  unsigned short* Qt  = (unsigned short*)(ws + ((size_t)20 << 20)); // 8 MB
  unsigned short* Ab  = (unsigned short*)(ws + ((size_t)28 << 20)); // 8 MB

  const int nx = MTOT * DMODEL;     // 4,194,304
  const int nw = DMODEL * DMODEL;   // 1,048,576
  cvt_bf16_kernel<<<(nx + 255) / 256, 256, 0, stream>>>(x,  Xb,  nx);
  cvt_bf16_kernel<<<(nw + 255) / 256, 256, 0, stream>>>(Wq, Wqb, nw);
  cvt_bf16_kernel<<<(nw + 255) / 256, 256, 0, stream>>>(Wo, Wob, nw);

  qproj_kernel<<<dim3(MTOT / 128, DMODEL / 128), 256, 0, stream>>>(Xb, Wqb, Qh, Qt);

  const int nwaves = BATCH * NHEADS * (T_SEQ / 16);   // 4096
  attn_kernel<<<nwaves / 8, 256, 0, stream>>>(Qh, Qt, relk, Ab);

  oproj_kernel<<<dim3(MTOT / 128, DMODEL / 128), 256, 0, stream>>>(Ab, Wob, (float*)d_out);
}
